// ProteinFeatures_3925600108550
// MI455X (gfx1250) — matrix operand order repaired and checked by reading, not compiled, not run on hardware
//
#include <hip/hip_runtime.h>
#include <math.h>

#define BS 2
#define NN 2048
#define TOPK 48
#define NRBF 16
#define NPOS 16
#define MAXREL 32
#define EIN 416   // 16 + 25*16
#define EF 128
#define NWV 2     // waves per block in edge kernel

typedef _Float16 v16h __attribute__((ext_vector_type(16)));
typedef float    v8f  __attribute__((ext_vector_type(8)));

union H16 { v16h v; uint4 u[2]; };

// pair tables: atom order [N=0, Ca=1, C=2, O=3, Cb=4]; slot 0 is Ca-Ca
__device__ const int PA[25] = {1,0,2,3,4,1,1,1,1,0,0,0,4,4,3,0,2,3,4,2,3,4,2,3,2};
__device__ const int PB[25] = {1,0,2,3,4,0,2,3,4,2,3,4,2,3,2,1,1,1,1,0,0,0,4,4,3};

// ---------------- Cb construction ----------------
__global__ void cb_kernel(const float* __restrict__ X, float* __restrict__ Cb) {
  int i = blockIdx.x * blockDim.x + threadIdx.x;
  if (i >= BS * NN) return;
  const float* x = X + (size_t)i * 12;
  float nx = x[0], ny = x[1], nz = x[2];
  float cax = x[3], cay = x[4], caz = x[5];
  float cx = x[6], cy = x[7], cz = x[8];
  float bx = cax - nx, by = cay - ny, bz = caz - nz;   // b = Ca - N
  float ex = cx - cax, ey = cy - cay, ez = cz - caz;   // c = C - Ca
  float ax = by * ez - bz * ey;                        // a = cross(b,c)
  float ay = bz * ex - bx * ez;
  float az = bx * ey - by * ex;
  Cb[(size_t)i * 3 + 0] = -0.58273431f * ax + 0.56802827f * bx - 0.54067466f * ex + cax;
  Cb[(size_t)i * 3 + 1] = -0.58273431f * ay + 0.56802827f * by - 0.54067466f * ey + cay;
  Cb[(size_t)i * 3 + 2] = -0.58273431f * az + 0.56802827f * bz - 0.54067466f * ez + caz;
}

// ---------------- W_edge -> f16, transposed to N-major ----------------
__global__ void wt_kernel(const float* __restrict__ W, _Float16* __restrict__ Wt) {
  int t = blockIdx.x * blockDim.x + threadIdx.x;
  if (t >= EIN * EF) return;
  int n = t / EIN, k = t % EIN;
  Wt[t] = (_Float16)W[(size_t)k * EF + n];   // Wt[n][k]
}

// ---------------- top-48 neighbor selection ----------------
__global__ void topk_kernel(const float* __restrict__ X, const float* __restrict__ mask,
                            int* __restrict__ Eidx, float* __restrict__ idx_out) {
  __shared__ float dist[NN];
  __shared__ float sval[256];
  __shared__ int   sidx[256];
  const int row = blockIdx.x;          // b*NN + i
  const int b   = row / NN;
  const int tid = threadIdx.x;
  const float* ca = X + ((size_t)row * 4 + 1) * 3;
  const float cax = ca[0], cay = ca[1], caz = ca[2];
  const float mi = mask[row];
  const float* Xb = X + (size_t)b * NN * 12;
  const float* Mb = mask + (size_t)b * NN;

  float lmax = -1e30f;
  for (int j = tid; j < NN; j += 256) {
    const float* cj = Xb + (size_t)j * 12 + 3;
    float dx = cax - cj[0], dy = cay - cj[1], dz = caz - cj[2];
    float r = sqrtf(dx * dx + dy * dy + dz * dz + 1e-6f);
    float d = mi * Mb[j] * r;
    dist[j] = d;
    lmax = fmaxf(lmax, d);
  }
  sval[tid] = lmax;
  __syncthreads();
  for (int s = 128; s > 0; s >>= 1) {
    if (tid < s) sval[tid] = fmaxf(sval[tid], sval[tid + s]);
    __syncthreads();
  }
  const float dmax = sval[0];
  __syncthreads();
  for (int j = tid; j < NN; j += 256) {
    float m = mi * Mb[j];
    dist[j] += (1.0f - m) * dmax;      // D_adjust
  }
  __syncthreads();

  for (int k = 0; k < TOPK; k++) {
    float bv = 1e30f; int bi = 0x7fffffff;
    for (int j = tid; j < NN; j += 256) {
      float v = dist[j];
      if (v < bv || (v == bv && j < bi)) { bv = v; bi = j; }
    }
    sval[tid] = bv; sidx[tid] = bi;
    __syncthreads();
    for (int s = 128; s > 0; s >>= 1) {
      if (tid < s) {
        float v2 = sval[tid + s]; int i2 = sidx[tid + s];
        if (v2 < sval[tid] || (v2 == sval[tid] && i2 < sidx[tid])) { sval[tid] = v2; sidx[tid] = i2; }
      }
      __syncthreads();
    }
    if (tid == 0) {
      int e = sidx[0];
      Eidx[(size_t)row * TOPK + k] = e;
      idx_out[(size_t)row * TOPK + k] = (float)e;
      dist[e] = 1e30f;
    }
    __syncthreads();
  }
}

// ---------------- fused features + (416x128) WMMA GEMM + LayerNorm ----------------
__global__ void __launch_bounds__(NWV * 32)
edge_kernel(const float* __restrict__ X, const float* __restrict__ Cb,
            const int* __restrict__ Ridx, const int* __restrict__ chain,
            const float* __restrict__ Wpos, const float* __restrict__ bpos,
            const _Float16* __restrict__ Wt,
            const float* __restrict__ gamma, const float* __restrict__ beta,
            const int* __restrict__ Eidx, float* __restrict__ Eout) {
  __shared__ _Float16 feaS[NWV][16][EIN];   // A tiles, row-major f16
  __shared__ float    outS[NWV][16][EF];    // D staging for LayerNorm
  __shared__ float    jatS[NWV][16][16];    // 5 atoms x 3 (padded) per neighbor
  __shared__ float    iatS[NWV][16];        // 5 atoms x 3 of residue i
  __shared__ int      jidxS[NWV][16];
  __shared__ float    statS[NWV][16][2];    // mean, rstd

  const int tid  = threadIdx.x;
  const int w    = tid >> 5;
  const int lane = tid & 31;
  const int l15  = lane & 15;
  const int hs   = lane >> 4;               // lane half select

  const long edgeBase = (long)blockIdx.x * (NWV * 16) + (long)w * 16;
  const int b   = (int)(edgeBase / (NN * TOPK));
  const int rem = (int)(edgeBase % (NN * TOPK));
  const int i   = rem / TOPK;
  const int ko  = rem % TOPK;               // multiple of 16 (48 = 3*16)
  const int row = b * NN + i;

  // --- stage neighbor indices and i-atoms ---
  if (lane < 16) {
    jidxS[w][lane] = Eidx[(size_t)row * TOPK + ko + lane];
    if (lane < 15) {
      int a = lane / 3, c = lane % 3;
      iatS[w][lane] = (a < 4) ? X[((size_t)row * 4 + a) * 3 + c]
                              : Cb[(size_t)row * 3 + c];
    }
  }
  __syncthreads();

  // --- stage j-atoms ---
  for (int t = lane; t < 16 * 15; t += 32) {
    int e = t / 15, c = t % 15;
    int j = jidxS[w][e];
    int a = c / 3, cc = c % 3;
    size_t jr = (size_t)b * NN + j;
    jatS[w][e][c] = (a < 4) ? X[(jr * 4 + a) * 3 + cc] : Cb[jr * 3 + cc];
  }
  // --- positional features (cols 0..15): one-hot @ W_pos == row select ---
  const int Ri = Ridx[row];
  const int ci = chain[row];
  for (int t = lane; t < 256; t += 32) {
    int e = t >> 4, c = t & 15;
    int j = jidxS[w][e];
    int off = Ri - Ridx[b * NN + j];
    int ec = (ci == chain[b * NN + j]) ? 1 : 0;
    int d = off + MAXREL;
    d = d < 0 ? 0 : (d > 2 * MAXREL ? 2 * MAXREL : d);
    d = ec ? d : (2 * MAXREL + 1);
    feaS[w][e][c] = (_Float16)(Wpos[d * NPOS + c] + bpos[c]);
  }
  __syncthreads();

  // --- RBF features (cols 16..415): 25 pairs x 16 RBFs ---
  for (int t = lane; t < 16 * 25; t += 32) {
    int e = t / 25, p = t % 25;
    int ai = PA[p] * 3, aj = PB[p] * 3;
    float dx = iatS[w][ai + 0] - jatS[w][e][aj + 0];
    float dy = iatS[w][ai + 1] - jatS[w][e][aj + 1];
    float dz = iatS[w][ai + 2] - jatS[w][e][aj + 2];
    float d = sqrtf(dx * dx + dy * dy + dz * dz + 1e-6f);
#pragma unroll
    for (int r = 0; r < NRBF; r++) {
      float z = (d - (2.0f + (20.0f / 15.0f) * r)) * 0.8f;   // /sigma, sigma=1.25
      feaS[w][e][16 + p * 16 + r] = (_Float16)__expf(-z * z);
    }
  }
  __syncthreads();

  // --- WMMA: 16 rows x 128 cols, K = 416 = 13 x 32 ---
  // Keep the K loop rolled: caps VGPR pressure < 256 (no s_set_vgpr_msb,
  // high occupancy); the inner 8-tile unroll still clusters 16 global b128
  // loads into clauses ahead of the 8 wmma issues.
  v8f acc[8] = {};
#pragma unroll 1
  for (int kc = 0; kc < EIN; kc += 32) {
    // A fragment (16-bit A 16x32 layout): lanes>=16 shift K by +8
    H16 A;
    const _Float16* ap = &feaS[w][l15][kc + hs * 8];
    A.u[0] = *(const uint4*)ap;          // K kc+hs*8 .. +7
    A.u[1] = *(const uint4*)(ap + 16);   // K kc+16+hs*8 .. +7
#pragma unroll
    for (int nt = 0; nt < 8; nt++) {
      // B fragment (32x16): lane holds col n; lanes>=16 shift K by +8 (same K order as A)
      H16 Bm;
      const _Float16* bp = &Wt[(size_t)(nt * 16 + l15) * EIN + kc + hs * 8];
      Bm.u[0] = *(const uint4*)bp;
      Bm.u[1] = *(const uint4*)(bp + 16);
      acc[nt] = __builtin_amdgcn_wmma_f32_16x16x32_f16(
          false, A.v, false, Bm.v, (short)0, acc[nt], false, false);
    }
  }

  // --- stage D to LDS (C/D layout: VGPR r, lanes<16: M=r; lanes>=16: M=8+r) ---
#pragma unroll
  for (int nt = 0; nt < 8; nt++)
#pragma unroll
    for (int r = 0; r < 8; r++)
      outS[w][hs * 8 + r][nt * 16 + l15] = acc[nt][r];
  __syncthreads();

  // --- per-row mean / rstd over 128 ---
  if (lane < 16) {
    float s = 0.0f, s2 = 0.0f;
    for (int c = 0; c < EF; c++) {
      float v = outS[w][lane][c];
      s += v; s2 += v * v;
    }
    float mu = s * (1.0f / EF);
    float var = s2 * (1.0f / EF) - mu * mu;
    statS[w][lane][0] = mu;
    statS[w][lane][1] = rsqrtf(var + 1e-5f);
  }
  __syncthreads();

  // --- normalize + affine + store ---
  for (int r = 0; r < 16; r++) {
    float mu = statS[w][r][0], rs = statS[w][r][1];
    size_t go = (size_t)(edgeBase + r) * EF;
    for (int c = lane; c < EF; c += 32)
      Eout[go + c] = (outS[w][r][c] - mu) * rs * gamma[c] + beta[c];
  }
}

extern "C" void kernel_launch(void* const* d_in, const int* in_sizes, int n_in,
                              void* d_out, int out_size, void* d_ws, size_t ws_size,
                              hipStream_t stream) {
  const float* X     = (const float*)d_in[0];
  const float* mask  = (const float*)d_in[1];
  const int*   Ridx  = (const int*)d_in[2];
  const int*   chain = (const int*)d_in[3];
  const float* Wpos  = (const float*)d_in[4];
  const float* bpos  = (const float*)d_in[5];
  const float* Wedge = (const float*)d_in[6];
  const float* gamma = (const float*)d_in[7];
  const float* beta  = (const float*)d_in[8];

  float* Eout    = (float*)d_out;
  float* idx_out = Eout + (size_t)BS * NN * TOPK * EF;   // E_idx tail (float-cast)

  char* ws = (char*)d_ws;
  float*     Cb   = (float*)ws;                                       // BS*NN*3 f32
  int*       Eidx = (int*)(ws + (size_t)BS * NN * 3 * sizeof(float)); // BS*NN*TOPK i32
  _Float16*  Wt   = (_Float16*)(ws + (size_t)BS * NN * 3 * sizeof(float)
                                   + (size_t)BS * NN * TOPK * sizeof(int)); // EF*EIN f16

  cb_kernel<<<(BS * NN + 255) / 256, 256, 0, stream>>>(X, Cb);
  wt_kernel<<<(EIN * EF + 255) / 256, 256, 0, stream>>>(Wedge, Wt);
  topk_kernel<<<BS * NN, 256, 0, stream>>>(X, mask, Eidx, idx_out);

  const int totalEdges = BS * NN * TOPK;                 // 196608
  edge_kernel<<<totalEdges / (NWV * 16), NWV * 32, 0, stream>>>(
      X, Cb, Ridx, chain, Wpos, bpos, Wt, gamma, beta, Eidx, Eout);
}
